// encoder_74972949119213
// MI455X (gfx1250) — matrix operand order repaired and checked by reading, not compiled, not run on hardware
//
#include <hip/hip_runtime.h>
#include <hip/hip_bf16.h>

typedef __attribute__((ext_vector_type(16))) _Float16 v16h;
typedef __attribute__((ext_vector_type(8)))  float    v8f;

#define BB  4
#define LL  2048
#define HH  128
#define NHH 2
#define GG  10

// ---------------- WMMA fragment helpers (16-bit layouts per CDNA5 ISA 7.12.2) ----
// A (16x32, M x K): lane L -> row m = L%16, half = L/16.
//   element e: K = 16*(e>>3) + 8*half + (e&7)  => two contiguous 8-half chunks.
__device__ __forceinline__ v16h load_a16(const _Float16* p /*row ptr at k0*/, int hf) {
  union { float4 f4[2]; v16h h; } u;
  u.f4[0] = *reinterpret_cast<const float4*>(p + 8 * hf);
  u.f4[1] = *reinterpret_cast<const float4*>(p + 16 + 8 * hf);
  return u.h;
}
// B (32x16, K x N): lane L -> col n = L%16, half = L/16; element e: K = 16*(e>>3) + 8*half + (e&7).
// p points at the n-th "column vector" laid out contiguously in K (i.e. transposed
// weights WT[n*K + k], or a key row for Q@K^T).
__device__ __forceinline__ v16h load_b16(const _Float16* p, int hf) {
  union { float4 f4[2]; v16h h; } u;
  u.f4[0] = *reinterpret_cast<const float4*>(p + 8 * hf);
  u.f4[1] = *reinterpret_cast<const float4*>(p + 16 + 8 * hf);
  return u.h;
}
__device__ __forceinline__ v8f wmma_f16(v16h a, v16h b, v8f c) {
  return __builtin_amdgcn_wmma_f32_16x16x32_f16(false, a, false, b, (short)0, c,
                                                false, false);
}

// ---------------- embedding kernels --------------------------------------------
__global__ void embed_obs_kernel(const float* __restrict__ obs,
                                 const float* __restrict__ mobs,
                                 const float* __restrict__ w_o0, const float* __restrict__ b_o0,
                                 const float* __restrict__ w_o1, const float* __restrict__ b_o1,
                                 const float* __restrict__ w_cobs, const float* __restrict__ b_cobs,
                                 const float* __restrict__ w_kp, const float* __restrict__ b_kp,
                                 float* __restrict__ out32, _Float16* __restrict__ out16) {
  int bl = blockIdx.x;          // 0 .. B*LO-1
  int j  = threadIdx.x;         // 0 .. 127
  __shared__ float v[257];
  float t   = obs[bl * 3 + 0];
  float chf = obs[bl * 3 + 1];
  float xv  = obs[bl * 3 + 2];
  float mo  = mobs[bl];
  int ch = (int)chf; ch = ch < 0 ? 0 : (ch > 40 ? 40 : ch);
  float tf = (j == 0) ? (t * w_o0[0] + b_o0[0]) : __sinf(t * w_o1[j - 1] + b_o1[j - 1]);
  v[j]       = tf * mo;
  v[128 + j] = fmaxf(w_cobs[ch * 128 + j] + b_cobs[j], 0.f) * mo;
  if (j == 0) v[256] = xv * mo;
  __syncthreads();
  float acc = b_kp[j];
  for (int i = 0; i < 257; ++i) acc += v[i] * w_kp[i * 128 + j];
  out32[(long)bl * 128 + j] = acc;
  out16[(long)bl * 128 + j] = (_Float16)acc;
}

__global__ void embed_q_kernel(const float* __restrict__ xq,
                               const float* __restrict__ mq,
                               const float* __restrict__ w_q0, const float* __restrict__ b_q0,
                               const float* __restrict__ w_q1, const float* __restrict__ b_q1,
                               const float* __restrict__ w_cq, const float* __restrict__ b_cq,
                               const float* __restrict__ w_qp, const float* __restrict__ b_qp,
                               float* __restrict__ out32, _Float16* __restrict__ out16) {
  int bl = blockIdx.x;
  int j  = threadIdx.x;
  __shared__ float v[256];
  float t   = xq[bl * 2 + 0];
  float chf = xq[bl * 2 + 1];
  float m   = mq[bl];
  int ch = (int)chf; ch = ch < 0 ? 0 : (ch > 40 ? 40 : ch);
  float tf = (j == 0) ? (t * w_q0[0] + b_q0[0]) : __sinf(t * w_q1[j - 1] + b_q1[j - 1]);
  v[j]       = tf * m;
  v[128 + j] = fmaxf(w_cq[ch * 128 + j] + b_cq[j], 0.f) * m;
  __syncthreads();
  float acc = b_qp[j];
  for (int i = 0; i < 256; ++i) acc += v[i] * w_qp[i * 128 + j];
  out32[(long)bl * 128 + j] = acc;
  out16[(long)bl * 128 + j] = (_Float16)acc;
}

// ---------------- weight convert + transpose (f32 [K,N] -> f16 [N,K]) -----------
__global__ void wT_kernel(const float* __restrict__ W, _Float16* __restrict__ WT,
                          int K, int N) {
  int idx = blockIdx.x * blockDim.x + threadIdx.x;
  if (idx >= K * N) return;
  int k = idx / N, n = idx % N;
  WT[(long)n * K + k] = (_Float16)W[(long)k * N + n];
}

// ---------------- generic 1-wave WMMA GEMM: out = act(A @ W + bias + res) -------
// A: f16 [M, K] (lda), WT: f16 [N, K]. Tile: 16 rows x 64 cols per wave.
__global__ __launch_bounds__(32)
void gemm16_kernel(const _Float16* __restrict__ A, int lda,
                   const _Float16* __restrict__ WT, int K,
                   const float* __restrict__ bias,
                   const float* __restrict__ res, int ldr,
                   float* __restrict__ out32, _Float16* __restrict__ out16, int ldo,
                   int relu, int mode, float* __restrict__ doutp) {
  const int lane = threadIdx.x;
  const int m16 = lane & 15, hf = lane >> 4;
  const int col0 = blockIdx.x * 64;
  const int row0 = blockIdx.y * 16;
  v8f acc[4] = {};
  const _Float16* arow = A + (long)(row0 + m16) * lda;
  for (int k0 = 0; k0 < K; k0 += 32) {
    v16h a = load_a16(arow + k0, hf);
    v16h bfr[4];
#pragma unroll
    for (int ct = 0; ct < 4; ++ct) {
      int col = col0 + ct * 16 + m16;
      bfr[ct] = load_b16(WT + (long)col * K + k0, hf);
    }
#pragma unroll
    for (int ct = 0; ct < 4; ++ct) acc[ct] = wmma_f16(a, bfr[ct], acc[ct]);
  }
#pragma unroll
  for (int ct = 0; ct < 4; ++ct) {
    int col = col0 + ct * 16 + m16;
    float bv = bias[col];
#pragma unroll
    for (int r = 0; r < 8; ++r) {
      int row = row0 + r + 8 * hf;
      float v = acc[ct][r] + bv;
      if (res) v += res[(long)row * ldr + col];
      if (relu) v = fmaxf(v, 0.f);
      if (mode == 0) {
        if (out32) out32[(long)row * ldo + col] = v;
        if (out16) out16[(long)row * ldo + col] = (_Float16)v;
      } else {
        // split-store: row = b*2048 + l ; col = g*128 + h ; -> [b][g][l][h]
        int b = row >> 11, l = row & 2047;
        int g = col >> 7, hc = col & 127;
        doutp[(((long)b * GG + g) * LL + l) * HH + hc] = v;
      }
    }
  }
}

// ---------------- two-pass flash attention (1 wave = 16 queries, 1 head) --------
__device__ __forceinline__ v8f score_tile(const _Float16* kvb, int kt,
                                          const v16h aq[4], int m16, int hf) {
  const _Float16* krow = kvb + (long)(kt * 16 + m16) * 256;
  v16h bk[4];
#pragma unroll
  for (int c4 = 0; c4 < 4; ++c4) bk[c4] = load_b16(krow + c4 * 32, hf);
  v8f c = {};
#pragma unroll
  for (int c4 = 0; c4 < 4; ++c4) c = wmma_f16(aq[c4], bk[c4], c);
  return c;
}

__global__ __launch_bounds__(32)
void attn_kernel(const _Float16* __restrict__ qp, const _Float16* __restrict__ kvp,
                 const float* __restrict__ maskQ, const float* __restrict__ maskK,
                 _Float16* __restrict__ outp, int Lq,
                 long qbs, long kvbs, long outbs, int nQt) {
  const int lane = threadIdx.x;
  const int m16 = lane & 15, hf = lane >> 4;
  const int qt = blockIdx.x % nQt;
  const int h  = (blockIdx.x / nQt) & (NHH - 1);
  const int b  = blockIdx.x / (nQt * NHH);
  const _Float16* qbase = qp + (long)b * qbs + h * HH;
  const _Float16* kvb   = kvp + (long)b * kvbs + h * HH;

  int qrow = qt * 16 + m16; if (qrow >= Lq) qrow = Lq - 1;
  v16h aq[4];
#pragma unroll
  for (int c = 0; c < 4; ++c) aq[c] = load_a16(qbase + (long)qrow * 256 + c * 32, hf);

  float mqv[8];
#pragma unroll
  for (int r = 0; r < 8; ++r) {
    int row = qt * 16 + r + 8 * hf;
    int rr = row < Lq ? row : Lq - 1;
    mqv[r] = maskQ ? maskQ[(long)b * Lq + rr] : 1.f;
  }
  const float scale = 0.08838834764831845f;  // 1/sqrt(128)
  float mrow[8], lsum[8];
#pragma unroll
  for (int r = 0; r < 8; ++r) { mrow[r] = -3.0e38f; lsum[r] = 0.f; }

  // ---- pass 1: streaming row max / sum-of-exp ----
  for (int kt = 0; kt < LL / 16; ++kt) {
    v8f s = score_tile(kvb, kt, aq, m16, hf);
    float mk = maskK[(long)b * LL + kt * 16 + m16];
#pragma unroll
    for (int r = 0; r < 8; ++r) {
      float sv = s[r] * scale;
      if (mk == 0.f || mqv[r] == 0.f) sv = -1.0e6f;
      float tmax = sv;
      for (int o = 1; o < 16; o <<= 1) tmax = fmaxf(tmax, __shfl_xor(tmax, o, 32));
      float nm = fmaxf(mrow[r], tmax);
      float ex = __expf(sv - nm);
      float ts = ex;
      for (int o = 1; o < 16; o <<= 1) ts += __shfl_xor(ts, o, 32);
      lsum[r] = lsum[r] * __expf(mrow[r] - nm) + ts;
      mrow[r] = nm;
    }
  }
  float invl[8];
#pragma unroll
  for (int r = 0; r < 8; ++r) invl[r] = 1.f / lsum[r];

  // ---- pass 2: P@V with LDS relayout (C-layout -> A-layout) ----
  __shared__ _Float16 pbuf[16 * 32];
  v8f acco[8] = {};
  for (int st = 0; st < LL / 32; ++st) {
#pragma unroll
    for (int s2 = 0; s2 < 2; ++s2) {
      int kt = st * 2 + s2;
      v8f s = score_tile(kvb, kt, aq, m16, hf);
      float mk = maskK[(long)b * LL + kt * 16 + m16];
#pragma unroll
      for (int r = 0; r < 8; ++r) {
        float sv = s[r] * scale;
        if (mk == 0.f || mqv[r] == 0.f) sv = -1.0e6f;
        float p = __expf(sv - mrow[r]) * invl[r];
        pbuf[(r + 8 * hf) * 32 + s2 * 16 + m16] = (_Float16)p;
      }
    }
    __syncthreads();
    v16h ap = load_a16(pbuf + m16 * 32, hf);
    // preload all 8 V fragments (strided gathers), then burst 8 WMMAs
    v16h vfr[8];
#pragma unroll
    for (int ct = 0; ct < 8; ++ct) {
      union { v16h h; _Float16 e[16]; } vb;
      const _Float16* vbase = kvb + (long)(st * 32 + 8 * hf) * 256 + ct * 16 + m16;
#pragma unroll
      for (int e = 0; e < 16; ++e) vb.e[e] = vbase[(long)(e + 8 * (e >> 3)) * 256];
      vfr[ct] = vb.h;
    }
#pragma unroll
    for (int ct = 0; ct < 8; ++ct) acco[ct] = wmma_f16(ap, vfr[ct], acco[ct]);
    __syncthreads();
  }
  // ---- store ----
#pragma unroll
  for (int ct = 0; ct < 8; ++ct)
#pragma unroll
    for (int r = 0; r < 8; ++r) {
      int row = qt * 16 + r + 8 * hf;
      if (row < Lq)
        outp[(long)b * outbs + (long)row * 256 + h * HH + ct * 16 + m16] =
            (_Float16)acco[ct][r];
    }
}

// ---------------- tiny scalar kernels for the G=10 mix path ---------------------
__global__ void qw_kernel(const float* __restrict__ mix_wt, const float* __restrict__ ww_q,
                          const float* __restrict__ ww_qb, _Float16* __restrict__ qw) {
  int g = blockIdx.x, j = threadIdx.x;  // 10 x 256
  float a = ww_qb[j];
  for (int k = 0; k < 128; ++k) a += mix_wt[g * 128 + k] * ww_q[k * 256 + j];
  qw[g * 256 + j] = (_Float16)a;
}

__global__ void mwres_kernel(const _Float16* __restrict__ attn_w,
                             const float* __restrict__ ww_o, const float* __restrict__ ww_ob,
                             const float* __restrict__ mix_wt, float* __restrict__ mw_) {
  int bg = blockIdx.x, j = threadIdx.x;  // (B*G) x 128
  int g = bg % GG, b = bg / GG;
  float a = ww_ob[j] + mix_wt[g * 128 + j];
  const _Float16* row = attn_w + (long)(b * 16 + g) * 256;
  for (int k = 0; k < 256; ++k) a += (float)row[k] * ww_o[k * 128 + j];
  mw_[(long)bg * 128 + j] = a;
}

__global__ void mix_kernel(const float* __restrict__ mw_, const float* __restrict__ w_mix,
                           const float* __restrict__ b_mix, float* __restrict__ out) {
  int b = blockIdx.x, t = threadIdx.x;  // B blocks x 32 threads
  __shared__ float lg[GG];
  if (t < GG) {
    float a = b_mix[0];
    const float* p = mw_ + (long)(b * GG + t) * 128;
    for (int j = 0; j < 128; ++j) a += p[j] * w_mix[j];
    lg[t] = a;
  }
  __syncthreads();
  if (t == 0) {
    float mx = -3.0e38f;
    for (int g = 0; g < GG; ++g) mx = fmaxf(mx, lg[g]);
    float e[GG], s = 0.f;
    for (int g = 0; g < GG; ++g) { e[g] = __expf(lg[g] - mx); s += e[g]; }
    for (int g = 0; g < GG; ++g) out[b * GG + g] = e[g] / s;
  }
}

// ---------------- launch --------------------------------------------------------
extern "C" void kernel_launch(void* const* d_in, const int* in_sizes, int n_in,
                              void* d_out, int out_size, void* d_ws, size_t ws_size,
                              hipStream_t stream) {
  const float* obs    = (const float*)d_in[0];
  const float* mobs   = (const float*)d_in[1];
  const float* xq     = (const float*)d_in[2];
  const float* mq     = (const float*)d_in[3];
  const float* w_q0   = (const float*)d_in[4];
  const float* b_q0   = (const float*)d_in[5];
  const float* w_q1   = (const float*)d_in[6];
  const float* b_q1   = (const float*)d_in[7];
  const float* w_o0   = (const float*)d_in[8];
  const float* b_o0   = (const float*)d_in[9];
  const float* w_o1   = (const float*)d_in[10];
  const float* b_o1   = (const float*)d_in[11];
  const float* w_cobs = (const float*)d_in[12];
  const float* b_cobs = (const float*)d_in[13];
  const float* w_cq   = (const float*)d_in[14];
  const float* b_cq   = (const float*)d_in[15];
  const float* w_qp   = (const float*)d_in[16];
  const float* b_qp   = (const float*)d_in[17];
  const float* w_kp   = (const float*)d_in[18];
  const float* b_kp   = (const float*)d_in[19];
  const float* ws_q   = (const float*)d_in[20];
  const float* ws_qb  = (const float*)d_in[21];
  const float* ws_o   = (const float*)d_in[22];
  const float* ws_ob  = (const float*)d_in[23];
  const float* ww_q   = (const float*)d_in[24];
  const float* ww_qb  = (const float*)d_in[25];
  const float* ww_o   = (const float*)d_in[26];
  const float* ww_ob  = (const float*)d_in[27];
  const float* wc_q   = (const float*)d_in[28];
  const float* wc_qb  = (const float*)d_in[29];
  const float* wc_o   = (const float*)d_in[30];
  const float* wc_ob  = (const float*)d_in[31];
  const float* w_split= (const float*)d_in[32];
  const float* b_split= (const float*)d_in[33];
  const float* mix_wt = (const float*)d_in[34];
  const float* w_mix  = (const float*)d_in[35];
  const float* b_mix  = (const float*)d_in[36];

  const long M = (long)BB * LL;  // 8192

  size_t off = 0;
  auto alloc = [&](size_t bytes) -> void* {
    void* p = (char*)d_ws + off;
    off += (bytes + 255) & ~(size_t)255;
    return p;
  };
  float*    obs_e32 = (float*)alloc(M * HH * 4);
  float*    qry_e32 = (float*)alloc(M * HH * 4);
  _Float16* obs_e16 = (_Float16*)alloc(M * HH * 2);
  _Float16* qry_e16 = (_Float16*)alloc(M * HH * 2);
  _Float16* proj_s  = (_Float16*)alloc(M * 256 * 2);
  _Float16* attn_s  = (_Float16*)alloc(M * 256 * 2);
  float*    x32     = (float*)alloc(M * HH * 4);
  _Float16* x16     = (_Float16*)alloc(M * HH * 2);
  _Float16* proj_w  = (_Float16*)alloc(M * 256 * 2);
  _Float16* proj_cq = (_Float16*)alloc(M * 256 * 2);
  _Float16* proj_ckv= (_Float16*)alloc(M * 256 * 2);
  _Float16* attn_c  = (_Float16*)alloc(M * 256 * 2);
  _Float16* xq2     = (_Float16*)alloc(M * HH * 2);
  _Float16* qw16    = (_Float16*)alloc(GG * 256 * 2);
  _Float16* attn_w  = (_Float16*)alloc(BB * 16 * 256 * 2);
  float*    mwbuf   = (float*)alloc(BB * GG * 128 * 4);
  _Float16* wsqT    = (_Float16*)alloc(128 * 256 * 2);
  _Float16* wsoT    = (_Float16*)alloc(256 * 128 * 2);
  _Float16* wwqT    = (_Float16*)alloc(128 * 256 * 2);
  _Float16* wcqT    = (_Float16*)alloc(128 * 256 * 2);
  _Float16* wcoT    = (_Float16*)alloc(256 * 128 * 2);
  _Float16* wspT    = (_Float16*)alloc(128 * 1280 * 2);

  float* dout = (float*)d_out;
  float* dout_mw = dout + (long)BB * GG * LL * HH;

  // weight transposes -> f16 [N,K]
  wT_kernel<<<(128 * 256 + 255) / 256, 256, 0, stream>>>(ws_q, wsqT, 128, 256);
  wT_kernel<<<(256 * 128 + 255) / 256, 256, 0, stream>>>(ws_o, wsoT, 256, 128);
  wT_kernel<<<(128 * 256 + 255) / 256, 256, 0, stream>>>(ww_q, wwqT, 128, 256);
  wT_kernel<<<(128 * 256 + 255) / 256, 256, 0, stream>>>(wc_q, wcqT, 128, 256);
  wT_kernel<<<(256 * 128 + 255) / 256, 256, 0, stream>>>(wc_o, wcoT, 256, 128);
  wT_kernel<<<(128 * 1280 + 255) / 256, 256, 0, stream>>>(w_split, wspT, 128, 1280);

  // embeds
  embed_obs_kernel<<<(int)M, 128, 0, stream>>>(obs, mobs, w_o0, b_o0, w_o1, b_o1,
                                               w_cobs, b_cobs, w_kp, b_kp,
                                               obs_e32, obs_e16);
  embed_q_kernel<<<(int)M, 128, 0, stream>>>(xq, mq, w_q0, b_q0, w_q1, b_q1,
                                             w_cq, b_cq, w_qp, b_qp,
                                             qry_e32, qry_e16);

  dim3 gN256(256 / 64, M / 16);
  dim3 gN128(128 / 64, M / 16);
  dim3 gN1280(1280 / 64, M / 16);

  // --- self attention block ---
  gemm16_kernel<<<gN256, 32, 0, stream>>>(obs_e16, 128, wsqT, 128, ws_qb,
                                          nullptr, 0, nullptr, proj_s, 256, 0, 0, nullptr);
  attn_kernel<<<BB * NHH * (LL / 16), 32, 0, stream>>>(proj_s, proj_s, mobs, mobs,
      attn_s, LL, (long)LL * 256, (long)LL * 256, (long)LL * 256, LL / 16);
  gemm16_kernel<<<gN128, 32, 0, stream>>>(attn_s, 256, wsoT, 256, ws_ob,
                                          obs_e32, 128, x32, x16, 128, 1, 0, nullptr);

  // --- mix-weights attention (G=10 queries, shared across batch) ---
  gemm16_kernel<<<gN256, 32, 0, stream>>>(x16, 128, wwqT, 128, ww_qb,
                                          nullptr, 0, nullptr, proj_w, 256, 0, 0, nullptr);
  qw_kernel<<<GG, 256, 0, stream>>>(mix_wt, ww_q, ww_qb, qw16);
  attn_kernel<<<BB * NHH, 32, 0, stream>>>(qw16, proj_w, nullptr, mobs,
      attn_w, GG, 0L, (long)LL * 256, (long)16 * 256, 1);
  mwres_kernel<<<BB * GG, 128, 0, stream>>>(attn_w, ww_o, ww_ob, mix_wt, mwbuf);
  mix_kernel<<<BB, 32, 0, stream>>>(mwbuf, w_mix, b_mix, dout_mw);

  // --- cross attention block ---
  gemm16_kernel<<<gN256, 32, 0, stream>>>(qry_e16, 128, wcqT, 128, wc_qb,
                                          nullptr, 0, nullptr, proj_cq, 256, 0, 0, nullptr);
  gemm16_kernel<<<gN256, 32, 0, stream>>>(x16, 128, wcqT, 128, wc_qb,
                                          nullptr, 0, nullptr, proj_ckv, 256, 0, 0, nullptr);
  attn_kernel<<<BB * NHH * (LL / 16), 32, 0, stream>>>(proj_cq, proj_ckv, mq, mobs,
      attn_c, LL, (long)LL * 256, (long)LL * 256, (long)LL * 256, LL / 16);
  gemm16_kernel<<<gN128, 32, 0, stream>>>(attn_c, 256, wcoT, 256, wc_ob,
                                          qry_e32, 128, nullptr, xq2, 128, 1, 0, nullptr);

  // --- split projection with transposed store into d_out ---
  gemm16_kernel<<<gN1280, 32, 0, stream>>>(xq2, 128, wspT, 128, b_split,
                                           nullptr, 0, nullptr, nullptr, 0, 0, 1, dout);
  (void)in_sizes; (void)n_in; (void)out_size; (void)ws_size;
}
